// GPT2_63264868270316
// MI455X (gfx1250) — matrix operand order repaired and checked by reading, not compiled, not run on hardware
//
#include <hip/hip_runtime.h>
#include <hip/hip_bf16.h>
#include <math.h>

// ---------------- constants (GPT-2 small, B=2 T=1024) ----------------
#define GV 50257
#define GD 1024
#define GH 16
#define GL 12
#define GB 2
#define GT 1024
#define GDH 64
#define GM (GB * GT)          // 2048 rows of activations

typedef __attribute__((ext_vector_type(16))) _Float16 v16h;
typedef __attribute__((ext_vector_type(8)))  float    v8f;

// ---------------- helpers ----------------
__device__ __forceinline__ unsigned short f2h(float f) {
    _Float16 h = (_Float16)f;
    unsigned short s;
    __builtin_memcpy(&s, &h, 2);
    return s;
}

__device__ __forceinline__ float gelu_tanh(float x) {
    float x3 = x * x * x;
    return 0.5f * x * (1.0f + tanhf(0.7978845608028654f * (x + 0.044715f * x3)));
}

union FragU { v16h v; unsigned int u[8]; };

// A-matrix 16xK fragment from LDS (row-major f16, 'stride' halfs per row).
// Per ISA 7.12.2: lane holds row m=lane&15; VGPR j holds K = (j>>2)*16 + hi*8 + (j&3)*2 (+koff).
__device__ __forceinline__ v16h frag_A(const unsigned short* base, int row, int stride,
                                       int koff, int lane) {
    FragU f;
    int m  = lane & 15;
    int hi = (lane >> 4) & 1;
    const unsigned short* r = base + (row + m) * stride;
#pragma unroll
    for (int j = 0; j < 8; ++j) {
        int k = koff + (((j >> 2) << 4) | (hi << 3) | ((j & 3) << 1));
        f.u[j] = *(const unsigned int*)(r + k);
    }
    return f.v;
}

// B-matrix Kx16 fragment from LDS stored N-major (Bt[n][k], 'stride' halfs per n-row).
// Lane holds col n=lane&15; VGPR j holds K = (j>>2)*16 + hi*8 + (j&3)*2 (+koff).
__device__ __forceinline__ v16h frag_B(const unsigned short* base, int col, int stride,
                                       int koff, int lane) {
    FragU f;
    int n  = lane & 15;
    int hi = (lane >> 4) & 1;
    const unsigned short* r = base + (col + n) * stride;
#pragma unroll
    for (int j = 0; j < 8; ++j) {
        int k = koff + (((j >> 2) << 4) | (hi << 3) | ((j & 3) << 1));
        f.u[j] = *(const unsigned int*)(r + k);
    }
    return f.v;
}

__device__ __forceinline__ v8f wmma16(v16h a, v16h b, v8f c) {
    return __builtin_amdgcn_wmma_f32_16x16x32_f16(false, a, false, b, (short)0, c,
                                                  false, false);
}

// ---------------- embedding: x = wte[inp] + wpe[t] ----------------
__global__ __launch_bounds__(256) void gpt2_embed(const int* __restrict__ inp,
                                                  const float* __restrict__ wte,
                                                  const float* __restrict__ wpe,
                                                  float* __restrict__ x) {
    int bt = blockIdx.x;            // 0..GM-1
    int t  = bt & (GT - 1);
    int id = inp[bt];
    const float* we = wte + (size_t)id * GD;
    const float* pe = wpe + (size_t)t * GD;
    float* xr = x + (size_t)bt * GD;
    for (int c = threadIdx.x; c < GD; c += 256) xr[c] = we[c] + pe[c];
}

// ---------------- layernorm: y = (x-mu)*rstd*g + b ----------------
__global__ __launch_bounds__(256) void gpt2_ln(const float* __restrict__ x,
                                               const float* __restrict__ g,
                                               const float* __restrict__ b,
                                               float* __restrict__ y) {
    __shared__ float s1[8], s2[8];
    int row = blockIdx.x;
    const float* xr = x + (size_t)row * GD;
    float vals[4], lsum = 0.f, lsq = 0.f;
#pragma unroll
    for (int i = 0; i < 4; ++i) {
        float v = xr[threadIdx.x + i * 256];
        vals[i] = v; lsum += v; lsq += v * v;
    }
#pragma unroll
    for (int o = 16; o > 0; o >>= 1) {
        lsum += __shfl_xor(lsum, o, 32);
        lsq  += __shfl_xor(lsq, o, 32);
    }
    int w = threadIdx.x >> 5, l = threadIdx.x & 31;
    if (l == 0) { s1[w] = lsum; s2[w] = lsq; }
    __syncthreads();
    if (threadIdx.x == 0) {
        float a = 0.f, c = 0.f;
#pragma unroll
        for (int i = 0; i < 8; ++i) { a += s1[i]; c += s2[i]; }
        s1[0] = a; s2[0] = c;
    }
    __syncthreads();
    float mean = s1[0] * (1.0f / GD);
    float var  = s2[0] * (1.0f / GD) - mean * mean;
    float rstd = rsqrtf(var + 1e-5f);
    float* yr = y + (size_t)row * GD;
#pragma unroll
    for (int i = 0; i < 4; ++i) {
        int c = threadIdx.x + i * 256;
        yr[c] = (vals[i] - mean) * rstd * g[c] + b[c];
    }
}

// ---------------- generic WMMA GEMM (register-staged ping-pong) ----------------
// C[M,N] = act( A[M,K] @ W + bias + resid ).  W is [K,N] (transB=0) or [N,K] (transB=1).
// Block tile 128x128, 8 waves each owning 64x32 (4 M-subtiles x 2 N-subtiles).
// Chunk c+1 is fetched into VGPRs while WMMAs consume LDS buffer c; one barrier/iter.
__global__ __launch_bounds__(256) void gpt2_gemm(const float* __restrict__ A,
                                                 const float* __restrict__ W,
                                                 const float* __restrict__ bias,
                                                 const float* __restrict__ resid,
                                                 float* __restrict__ C,
                                                 int M, int N, int K,
                                                 int transB, int act) {
    __shared__ unsigned short As[2][128 * 32];   // row-major [m][k]
    __shared__ unsigned short Bs[2][128 * 32];   // N-major  [n][k]
    const int m0 = blockIdx.y * 128, n0 = blockIdx.x * 128;
    const int tid = threadIdx.x, w = tid >> 5, lane = tid & 31;
    const int wm = (w & 1) * 64;        // wave M offset within block tile
    const int wn = (w >> 1) * 32;       // wave N offset within block tile

    float ar[16], br[16];               // register staging for next chunk

    auto fetchA = [&](int k0) {
#pragma unroll
        for (int i = 0; i < 16; ++i) {
            int e = i * 256 + tid, r = e >> 5, k = e & 31;      // coalesced over k
            ar[i] = A[(size_t)(m0 + r) * K + k0 + k];
        }
    };
    auto fetchB = [&](int k0) {
        if (!transB) {
#pragma unroll
            for (int i = 0; i < 16; ++i) {
                int e = i * 256 + tid, n = e & 127, k = e >> 7; // coalesced over n
                int gn = n0 + n;
                br[i] = (gn < N) ? W[(size_t)(k0 + k) * N + gn] : 0.f;
            }
        } else {
#pragma unroll
            for (int i = 0; i < 16; ++i) {
                int e = i * 256 + tid, k = e & 31, n = e >> 5;  // coalesced over k
                int gn = n0 + n;
                br[i] = (gn < N) ? W[(size_t)gn * K + k0 + k] : 0.f;
            }
        }
    };
    auto stash = [&](int buf) {
#pragma unroll
        for (int i = 0; i < 16; ++i) {
            int e = i * 256 + tid, r = e >> 5, k = e & 31;
            As[buf][r * 32 + k] = f2h(ar[i]);
        }
        if (!transB) {
#pragma unroll
            for (int i = 0; i < 16; ++i) {
                int e = i * 256 + tid, n = e & 127, k = e >> 7;
                Bs[buf][n * 32 + k] = f2h(br[i]);
            }
        } else {
#pragma unroll
            for (int i = 0; i < 16; ++i) {
                int e = i * 256 + tid, k = e & 31, n = e >> 5;
                Bs[buf][n * 32 + k] = f2h(br[i]);
            }
        }
    };

    v8f zero = {};
    v8f acc[4][2];
#pragma unroll
    for (int ms = 0; ms < 4; ++ms)
#pragma unroll
        for (int ns = 0; ns < 2; ++ns) acc[ms][ns] = zero;

    fetchA(0); fetchB(0); stash(0);
    __syncthreads();

    const int nchunks = K >> 5;
    for (int c = 0; c < nchunks; ++c) {
        const int buf = c & 1;
        if (c + 1 < nchunks) { fetchA((c + 1) << 5); fetchB((c + 1) << 5); }

        v16h bf[2];
#pragma unroll
        for (int ns = 0; ns < 2; ++ns) bf[ns] = frag_B(&Bs[buf][0], wn + ns * 16, 32, 0, lane);
#pragma unroll
        for (int ms = 0; ms < 4; ++ms) {
            v16h af = frag_A(&As[buf][0], wm + ms * 16, 32, 0, lane);
#pragma unroll
            for (int ns = 0; ns < 2; ++ns) acc[ms][ns] = wmma16(af, bf[ns], acc[ms][ns]);
        }

        if (c + 1 < nchunks) stash(buf ^ 1);   // writes opposite buffer: no race with
                                               // current readers; prior readers of it
                                               // finished before last barrier
        __syncthreads();                       // stores of c+1 visible before its reads
    }

    // epilogue: C layout — VGPR j <-> row j + 8*(lane>=16); col = lane&15
    const int hi = lane >> 4, col = lane & 15;
#pragma unroll
    for (int ms = 0; ms < 4; ++ms)
#pragma unroll
        for (int ns = 0; ns < 2; ++ns) {
            int gn = n0 + wn + ns * 16 + col;
            if (gn >= N) continue;
            float bv = bias ? bias[gn] : 0.f;
#pragma unroll
            for (int j = 0; j < 8; ++j) {
                int gm = m0 + wm + ms * 16 + j + 8 * hi;
                float v = acc[ms][ns][j] + bv;
                if (resid) v += resid[(size_t)gm * N + gn];
                if (act == 1) v = gelu_tanh(v);
                C[(size_t)gm * N + gn] = v;
            }
        }
}

// ---------------- fused flash attention (un-masked, matching reference) --------
// grid: (T/128, H, B), block 256 = 8 waves; each wave owns 16 query rows.
__global__ __launch_bounds__(256) void gpt2_attn(const float* __restrict__ qkv,
                                                 float* __restrict__ ctx) {
    __shared__ unsigned short Ks[64 * 64];      // K chunk, [key][dh]
    __shared__ unsigned short Vt[64 * 64];      // V chunk transposed, [dh][key]
    __shared__ unsigned short Ps[8 * 16 * 64];  // per-wave Q-then-P staging
    const int b = blockIdx.z, hh = blockIdx.y, qb = blockIdx.x * 128;
    const int tid = threadIdx.x, w = tid >> 5, lane = tid & 31;
    const int hi = lane >> 4, ln15 = lane & 15;
    unsigned short* Pw = Ps + w * (16 * 64);

    const size_t rstride = 3 * GD;                       // qkv row stride
    const size_t base = (size_t)b * GT * rstride + (size_t)hh * GDH;

    // stage this wave's 16x64 Q strip (fp32 -> f16)
    for (int e = lane; e < 16 * 64; e += 32) {
        int r = e >> 6, c = e & 63;
        Pw[r * 64 + c] = f2h(qkv[base + (size_t)(qb + w * 16 + r) * rstride + c]);
    }
    v16h qf0 = frag_A(Pw, 0, 64, 0, lane);
    v16h qf1 = frag_A(Pw, 0, 64, 32, lane);

    v8f zero = {};
    v8f O[4];
#pragma unroll
    for (int nt = 0; nt < 4; ++nt) O[nt] = zero;
    float mrow[8], lrow[8];
#pragma unroll
    for (int j = 0; j < 8; ++j) { mrow[j] = -1e30f; lrow[j] = 0.f; }
    const float scale = 0.125f;                 // 1/sqrt(64)

    for (int kc = 0; kc < GT / 64; ++kc) {
        const int kb = kc * 64;
        // stage K chunk row-major and V chunk transposed
        for (int e = tid; e < 4096; e += 256) {
            int r = e >> 6, c = e & 63;
            size_t rowoff = base + (size_t)(kb + r) * rstride;
            Ks[r * 64 + c] = f2h(qkv[rowoff + GD + c]);
            Vt[c * 64 + r] = f2h(qkv[rowoff + 2 * GD + c]);
        }
        __syncthreads();

        // S = Q @ K^T  (B fragment: Bt[n=key][k=dh] == Ks row-major)
        v8f S[4];
#pragma unroll
        for (int nt = 0; nt < 4; ++nt) S[nt] = zero;
#pragma unroll
        for (int cc = 0; cc < 2; ++cc) {
            v16h qf = cc ? qf1 : qf0;
#pragma unroll
            for (int nt = 0; nt < 4; ++nt) {
                v16h kf = frag_B(Ks, nt * 16, 64, cc * 32, lane);
                S[nt] = wmma16(qf, kf, S[nt]);
            }
        }

        // online softmax per accumulator row (row = j + 8*hi; cols across 16 lanes)
#pragma unroll
        for (int j = 0; j < 8; ++j) {
            float mx = S[0][j];
            mx = fmaxf(mx, S[1][j]); mx = fmaxf(mx, S[2][j]); mx = fmaxf(mx, S[3][j]);
            mx *= scale;
#pragma unroll
            for (int msk = 1; msk < 16; msk <<= 1) mx = fmaxf(mx, __shfl_xor(mx, msk, 32));
            float mnew  = fmaxf(mrow[j], mx);
            float alpha = __expf(mrow[j] - mnew);
            float rs = 0.f;
#pragma unroll
            for (int nt = 0; nt < 4; ++nt) {
                float p = __expf(S[nt][j] * scale - mnew);
                rs += p;
                Pw[(j + 8 * hi) * 64 + nt * 16 + ln15] = f2h(p);
            }
#pragma unroll
            for (int msk = 1; msk < 16; msk <<= 1) rs += __shfl_xor(rs, msk, 32);
            lrow[j] = lrow[j] * alpha + rs;
            mrow[j] = mnew;
#pragma unroll
            for (int nt = 0; nt < 4; ++nt) O[nt][j] *= alpha;
        }

        // O += P @ V  (A fragment from Pw; B fragment: Bt[n=dh][k=key] == Vt)
#pragma unroll
        for (int cc = 0; cc < 2; ++cc) {
            v16h pf = frag_A(Pw, 0, 64, cc * 32, lane);
#pragma unroll
            for (int nt = 0; nt < 4; ++nt) {
                v16h vf = frag_B(Vt, nt * 16, 64, cc * 32, lane);
                O[nt] = wmma16(pf, vf, O[nt]);
            }
        }
        __syncthreads();
    }

    // write ctx [B,T,D]
#pragma unroll
    for (int j = 0; j < 8; ++j) {
        float inv = 1.0f / lrow[j];
        int gr = qb + w * 16 + j + 8 * hi;
        float* dst = ctx + ((size_t)b * GT + gr) * GD + (size_t)hh * GDH;
#pragma unroll
        for (int nt = 0; nt < 4; ++nt) dst[nt * 16 + ln15] = O[nt][j] * inv;
    }
}

// ---------------- host driver ----------------
extern "C" void kernel_launch(void* const* d_in, const int* in_sizes, int n_in,
                              void* d_out, int out_size, void* d_ws, size_t ws_size,
                              hipStream_t stream) {
    (void)in_sizes; (void)n_in; (void)out_size; (void)ws_size;
    const int*   inp      = (const int*)  d_in[0];
    const float* wte      = (const float*)d_in[1];
    const float* wpe      = (const float*)d_in[2];
    const float* c_attn_w = (const float*)d_in[3];
    const float* c_attn_b = (const float*)d_in[4];
    const float* c_proj_w = (const float*)d_in[5];
    const float* c_proj_b = (const float*)d_in[6];
    const float* ln1_g    = (const float*)d_in[7];
    const float* ln1_b    = (const float*)d_in[8];
    const float* ln2_g    = (const float*)d_in[9];
    const float* ln2_b    = (const float*)d_in[10];
    const float* fc_w     = (const float*)d_in[11];
    const float* fc_b     = (const float*)d_in[12];
    const float* fcp_w    = (const float*)d_in[13];
    const float* fcp_b    = (const float*)d_in[14];
    const float* lnf_g    = (const float*)d_in[15];
    const float* lnf_b    = (const float*)d_in[16];
    float* out = (float*)d_out;

    float* ws  = (float*)d_ws;
    float* x    = ws;                       // GM*GD        = 2,097,152
    float* h    = ws + 2097152;             // GM*GD
    float* qkv  = ws + 4194304;             // GM*3*GD      = 6,291,456
    float* ctx  = ws + 10485760;            // GM*GD
    float* fc   = ws + 12582912;            // GM*4*GD      = 8,388,608

    dim3 blk(256);

    gpt2_embed<<<GM, blk, 0, stream>>>(inp, wte, wpe, x);

    for (int l = 0; l < GL; ++l) {
        gpt2_ln<<<GM, blk, 0, stream>>>(x, ln1_g + l * GD, ln1_b + l * GD, h);
        gpt2_gemm<<<dim3(24, 16), blk, 0, stream>>>(
            h, c_attn_w + (size_t)l * GD * 3 * GD, c_attn_b + (size_t)l * 3 * GD,
            nullptr, qkv, GM, 3 * GD, GD, 0, 0);
        gpt2_attn<<<dim3(GT / 128, GH, GB), blk, 0, stream>>>(qkv, ctx);
        gpt2_gemm<<<dim3(8, 16), blk, 0, stream>>>(
            ctx, c_proj_w + (size_t)l * GD * GD, c_proj_b + (size_t)l * GD,
            x, x, GM, GD, GD, 0, 0);
        gpt2_ln<<<GM, blk, 0, stream>>>(x, ln2_g + l * GD, ln2_b + l * GD, h);
        gpt2_gemm<<<dim3(32, 16), blk, 0, stream>>>(
            h, fc_w + (size_t)l * GD * 4 * GD, fc_b + (size_t)l * 4 * GD,
            nullptr, fc, GM, 4 * GD, GD, 0, 1);
        gpt2_gemm<<<dim3(8, 16), blk, 0, stream>>>(
            fc, fcp_w + (size_t)l * 4 * GD * GD, fcp_b + (size_t)l * GD,
            x, x, GM, GD, 4 * GD, 0, 0);
    }

    gpt2_ln<<<GM, blk, 0, stream>>>(x, lnf_g, lnf_b, h);
    // logits = h @ wte^T  (weight tying, transB): N = 50257
    gpt2_gemm<<<dim3((GV + 127) / 128, 16), blk, 0, stream>>>(
        h, wte, nullptr, nullptr, out, GM, GV, GD, 1, 0);
}
